// AttentionModule_44504451121345
// MI455X (gfx1250) — matrix operand order repaired and checked by reading, not compiled, not run on hardware
//
#include <hip/hip_runtime.h>

// CDNA5 / gfx1250 fused attention:
//   0) transpose+bf16-convert the 4 weight matrices (so WMMA B-fragments are contiguous)
//   1) QKV projection via v_wmma_f32_16x16x32_bf16 (Q pre-scaled by 1/sqrt(D); V stored transposed)
//   2) flash attention: K/V tiles DMA'd to LDS by the Tensor Data Mover (double-buffered,
//      TENSORcnt-tracked), online softmax, P transposed C->A layout via per-wave LDS + s_wait_dscnt
//   3) output projection + bias -> f32 out

typedef __attribute__((ext_vector_type(16))) __bf16 v16bf;
typedef __attribute__((ext_vector_type(8)))  __bf16 v8bf;
typedef __attribute__((ext_vector_type(8)))  float  v8f;
typedef __attribute__((ext_vector_type(4)))  unsigned int v4u;
typedef __attribute__((ext_vector_type(4)))  int v4i;
typedef __attribute__((ext_vector_type(8)))  int v8i;

#define DIM    256
#define SEQ    4096
#define NBATCH 4
#define MTOT   (NBATCH*SEQ)   // 16384 total rows

// LDS row strides (elements), padded for 32B-aligned v16bf loads + bank spread
#define LKS 272   // K tile: 32 rows x 256 cols  (row stride 544 B; 128 dwords + 8 dword pad)
#define LVS 48    // Vt tile: 256 rows x 32 cols (row stride 96 B; 16 dwords + 8 dword pad)
#define LPS 40    // P tile per wave: 16 x 32    (row stride 80 B, 16-aligned)

static __device__ __forceinline__ v8f wmma_bf16(v16bf a, v16bf b, v8f c) {
  return __builtin_amdgcn_wmma_f32_16x16x32_bf16(false, a, false, b, (short)0, c,
                                                 false, false);
}

// ---------------------------------------------------------------------------
// Tensor Data Mover: 2-D tile load, D# built per CDNA5 ISA
// ---------------------------------------------------------------------------
#if __has_builtin(__builtin_amdgcn_tensor_load_to_lds)
#define USE_TDM 1
// generic LDS pointer low 32 bits == wave-relative LDS byte address (flat aperture rule)
static __device__ __forceinline__ unsigned lds_byte_off(const void* p) {
  return (unsigned)(unsigned long long)p;
}
// cfg packs D# group1 dword0: data_size=2B | pad_enable | pad_interval | pad_amount
static __device__ __forceinline__ void tdm_load_2d(
    const void* gaddr, unsigned lds_addr, unsigned cfg,
    unsigned tdim0, unsigned tdim1, unsigned tile0, unsigned tile1,
    unsigned stride0)
{
  const unsigned long long ga = (unsigned long long)gaddr;
  v4u g0;
  g0[0] = 1u;                                              // count=1, user mode
  g0[1] = lds_addr;                                        // lds_addr[31:0]
  g0[2] = (unsigned)(ga & 0xffffffffu);                    // global_addr[31:0]
  g0[3] = (unsigned)((ga >> 32) & 0x01ffffffu) | 0x80000000u; // addr[56:32] | type=2
  v8i g1;
  g1[0] = (int)cfg;                                        // mask=0, data_size, pad cfg
  g1[1] = (int)((tdim0 & 0xffffu) << 16);                  // [63:48] tensor_dim0 lo
  g1[2] = (int)((tdim0 >> 16) | ((tdim1 & 0xffffu) << 16));// dim0 hi | dim1 lo
  g1[3] = (int)((tdim1 >> 16) | (tile0 << 16));            // dim1 hi | tile_dim0
  g1[4] = (int)tile1;                                      // tile_dim1, tile_dim2=0
  g1[5] = (int)stride0;                                    // tensor_dim0_stride lo
  g1[6] = 0;                                               // stride0 hi | dim1_stride lo
  g1[7] = 0;
  v4i z4 = {0, 0, 0, 0};
#if __clang_major__ >= 23
  v8i z8 = {0, 0, 0, 0, 0, 0, 0, 0};
  __builtin_amdgcn_tensor_load_to_lds(g0, g1, z4, z4, z8, 0);
#else
  __builtin_amdgcn_tensor_load_to_lds(g0, g1, z4, z4, 0);
#endif
}
// data_size=1 (2 bytes) | pad_enable | pad_interval | pad_amount(=7 -> 8 dwords = 16 elems)
#define TDM_KCFG ((1u << 16) | (1u << 20) | (6u << 22) | (7u << 25)) // pad every 128 dw
#define TDM_VCFG ((1u << 16) | (1u << 20) | (3u << 22) | (7u << 25)) // pad every 16 dw
#else
#define USE_TDM 0
#endif

// ---------------------------------------------------------------------------
// Kernel 0: W[k][n] f32 -> WT[n][k] bf16 for Wq,Wk,Wv,Wo (blockIdx.y selects)
// ---------------------------------------------------------------------------
__global__ __launch_bounds__(256) void transpose_weights(
    const float* __restrict__ Wq, const float* __restrict__ Wk,
    const float* __restrict__ Wv, const float* __restrict__ Wo,
    __bf16* __restrict__ WT)
{
  const float* W = (blockIdx.y == 0) ? Wq : (blockIdx.y == 1) ? Wk
                  : (blockIdx.y == 2) ? Wv : Wo;
  __bf16* T = WT + (size_t)blockIdx.y * (DIM * DIM);
  const int k = blockIdx.x;      // input row
  const int n = threadIdx.x;     // input col
  T[(size_t)n * DIM + k] = (__bf16)W[(size_t)k * DIM + n];
}

// ---------------------------------------------------------------------------
// Kernel 1: Q = (x@Wq + bq) * 1/16  (bf16), K = x@Wk + bk (bf16),
//           Vt[b][d][s] = (x@Wv + bv) (bf16, transposed)
// One wave = 16 rows; block = 8 waves = 128 rows; grid = 128 blocks.
// ---------------------------------------------------------------------------
__global__ __launch_bounds__(256) void qkv_proj_kernel(
    const float* __restrict__ x, const __bf16* __restrict__ WT,
    const float* __restrict__ bq, const float* __restrict__ bk,
    const float* __restrict__ bv,
    __bf16* __restrict__ Qb, __bf16* __restrict__ Kb, __bf16* __restrict__ Vt)
{
  const int lane = threadIdx.x & 31;
  const int wave = threadIdx.x >> 5;
  const int tl   = lane & 15;
  const int h    = lane >> 4;
  const int m0   = blockIdx.x * 128 + wave * 16;

  // warm L2 with the weights every block reuses (global_prefetch_b8)
  __builtin_prefetch(WT + (size_t)threadIdx.x * 256, 0, 0);
  __builtin_prefetch(WT + 65536 + (size_t)threadIdx.x * 256, 0, 0);
  __builtin_prefetch(WT + 131072 + (size_t)threadIdx.x * 256, 0, 0);

  const __bf16* WqT = WT;
  const __bf16* WkT = WT + 65536;
  const __bf16* WvT = WT + 131072;

  // A fragments: 16x32 bf16 tiles of x for this wave's 16 rows, all K
  v16bf a[8];
  const float* xr = x + (size_t)(m0 + tl) * DIM;
  #pragma unroll
  for (int kt = 0; kt < 8; ++kt) {
    const float* p0 = xr + kt * 32 + 16 * h;       // k = kt*32 + 16h + e   (e<8)
    const float* p1 = xr + kt * 32 + 16 * h + 8;   // k = kt*32 + 16h + e   (e>=8)
    float4 f0 = *(const float4*)(p0);
    float4 f1 = *(const float4*)(p0 + 4);
    float4 f2 = *(const float4*)(p1);
    float4 f3 = *(const float4*)(p1 + 4);
    float t[16] = {f0.x,f0.y,f0.z,f0.w, f1.x,f1.y,f1.z,f1.w,
                   f2.x,f2.y,f2.z,f2.w, f3.x,f3.y,f3.z,f3.w};
    #pragma unroll
    for (int e = 0; e < 16; ++e) a[kt][e] = (__bf16)t[e];
  }

  const int batch = m0 >> 12;
  const int sl    = m0 & (SEQ - 1);

  for (int nt = 0; nt < 16; ++nt) {
    v8f cq = {}; v8f ck = {}; v8f cv = {};
    const __bf16* wq = WqT + (size_t)(nt * 16 + tl) * DIM + 16 * h;
    const __bf16* wk = WkT + (size_t)(nt * 16 + tl) * DIM + 16 * h;
    const __bf16* wv = WvT + (size_t)(nt * 16 + tl) * DIM + 16 * h;
    #pragma unroll
    for (int kt = 0; kt < 8; ++kt) {
      v16bf bqf = *(const v16bf*)(wq + kt * 32);
      v16bf bkf = *(const v16bf*)(wk + kt * 32);
      v16bf bvf = *(const v16bf*)(wv + kt * 32);
      cq = wmma_bf16(a[kt], bqf, cq);
      ck = wmma_bf16(a[kt], bkf, ck);
      cv = wmma_bf16(a[kt], bvf, cv);
    }
    const int col = nt * 16 + tl;
    const float bqv = bq[col], bkv = bk[col], bvv = bv[col];
    #pragma unroll
    for (int r = 0; r < 8; ++r) {
      const int mrow = r + 8 * h;                    // C/D layout: M = r + 8*half
      const size_t s = (size_t)(m0 + mrow);
      Qb[s * DIM + col] = (__bf16)((cq[r] + bqv) * 0.0625f);  // fold 1/sqrt(256)
      Kb[s * DIM + col] = (__bf16)(ck[r] + bkv);
      Vt[(size_t)batch * (DIM * SEQ) + (size_t)col * SEQ + (sl + mrow)] =
          (__bf16)(cv[r] + bvv);
    }
  }
}

// ---------------------------------------------------------------------------
// Kernel 2: flash attention. One wave = 16 query rows, block = 8 waves.
// 32 keys/iter; K and Vt tiles land in LDS via TDM (double-buffered) and are
// shared by all waves of the block.
// ---------------------------------------------------------------------------
__global__ __launch_bounds__(256) void flash_attn_kernel(
    const __bf16* __restrict__ Qb, const __bf16* __restrict__ Kb,
    const __bf16* __restrict__ Vt, __bf16* __restrict__ Ob)
{
  __shared__ __bf16 ldsK[2][32 * LKS];     // 2 x 17408 B
  __shared__ __bf16 ldsVt[2][256 * LVS];   // 2 x 24576 B
  __shared__ __bf16 ldsP[8 * 16 * LPS];    // 10240 B

  const int tid  = threadIdx.x;
  const int lane = tid & 31;
  const int wave = tid >> 5;
  const int tl   = lane & 15;
  const int h    = lane >> 4;
  const int m0   = blockIdx.x * 128 + wave * 16;
  const int batch   = m0 >> 12;
  const int keyBase = batch << 12;

  const __bf16* Kbase = Kb + (size_t)keyBase * DIM;
  const __bf16* Vbase = Vt + (size_t)batch * (DIM * SEQ);

  // Q fragments (A layout, 16x32 per dk tile)
  v16bf q[8];
  const __bf16* qr = Qb + (size_t)(m0 + tl) * DIM;
  #pragma unroll
  for (int dk = 0; dk < 8; ++dk) {
    v8bf lo = *(const v8bf*)(qr + dk * 32 + 16 * h);
    v8bf hi = *(const v8bf*)(qr + dk * 32 + 16 * h + 8);
    #pragma unroll
    for (int e = 0; e < 8; ++e) { q[dk][e] = lo[e]; q[dk][8 + e] = hi[e]; }
  }

  v8f acc[16];
  v8f vzero = {};
  #pragma unroll
  for (int nt = 0; nt < 16; ++nt) acc[nt] = vzero;
  float rmax[8], rsum[8];
  #pragma unroll
  for (int r = 0; r < 8; ++r) { rmax[r] = -3.0e38f; rsum[r] = 0.0f; }

  __bf16* pb = ldsP + wave * 16 * LPS;

#if USE_TDM
  if (wave == 0) {   // one wave drives the TDM; EXEC is ignored by tensor ops
    tdm_load_2d(Kbase, lds_byte_off(&ldsK[0][0]), TDM_KCFG,
                DIM, SEQ, DIM, 32, DIM);
    tdm_load_2d(Vbase, lds_byte_off(&ldsVt[0][0]), TDM_VCFG,
                SEQ, DIM, 32, DIM, SEQ);
  }
#endif

  for (int t = 0; t < SEQ / 32; ++t) {
    const int k0  = t * 32;
    const int cur = t & 1;

#if USE_TDM
    if (wave == 0) __builtin_amdgcn_s_wait_tensorcnt(0);  // tile t landed in LDS
    __syncthreads();
    if (wave == 0 && k0 + 32 < SEQ) {                     // kick tile t+1 DMA
      const int nb = cur ^ 1;
      tdm_load_2d(Kbase + (size_t)(k0 + 32) * DIM, lds_byte_off(&ldsK[nb][0]),
                  TDM_KCFG, DIM, SEQ, DIM, 32, DIM);
      tdm_load_2d(Vbase + (k0 + 32), lds_byte_off(&ldsVt[nb][0]),
                  TDM_VCFG, SEQ, DIM, 32, DIM, SEQ);
    }
#else
    __syncthreads();
    {
      const uint4* srcK = (const uint4*)(Kbase + (size_t)k0 * DIM);
      #pragma unroll
      for (int i = 0; i < 4; ++i) {
        const int c = tid + i * 256;
        *(uint4*)(&ldsK[cur][0] + (c >> 5) * LKS + (c & 31) * 8) = srcK[c];
      }
      const uint4* srcV = (const uint4*)(Vbase + (size_t)tid * SEQ + k0);
      #pragma unroll
      for (int i = 0; i < 4; ++i)
        *(uint4*)(&ldsVt[cur][0] + tid * LVS + i * 8) = srcV[i];
    }
    __syncthreads();
#endif

    const __bf16* kbuf = &ldsK[cur][0];
    const __bf16* vbuf = &ldsVt[cur][0];

    // scores: S = Q @ K^T for 32 keys (two 16-col WMMA tiles)
    v8f s0 = {}; v8f s1 = {};
    #pragma unroll
    for (int dk = 0; dk < 8; ++dk) {
      v16bf b0 = *(const v16bf*)(kbuf + (size_t)tl * LKS + dk * 32 + 16 * h);
      v16bf b1 = *(const v16bf*)(kbuf + (size_t)(16 + tl) * LKS + dk * 32 + 16 * h);
      s0 = wmma_bf16(q[dk], b0, s0);
      s1 = wmma_bf16(q[dk], b1, s1);
    }

    // online softmax (row m = r + 8h lives in vgpr r, striped over 16 lanes)
    float alpha[8];
    #pragma unroll
    for (int r = 0; r < 8; ++r) {
      float v = fmaxf(s0[r], s1[r]);
      v = fmaxf(v, __shfl_xor(v, 1, 32));
      v = fmaxf(v, __shfl_xor(v, 2, 32));
      v = fmaxf(v, __shfl_xor(v, 4, 32));
      v = fmaxf(v, __shfl_xor(v, 8, 32));
      const float mn = fmaxf(rmax[r], v);
      alpha[r] = __expf(rmax[r] - mn);
      rmax[r] = mn;
      const float p0 = __expf(s0[r] - mn);
      const float p1 = __expf(s1[r] - mn);
      float ps = p0 + p1;
      ps += __shfl_xor(ps, 1, 32);
      ps += __shfl_xor(ps, 2, 32);
      ps += __shfl_xor(ps, 4, 32);
      ps += __shfl_xor(ps, 8, 32);
      rsum[r] = rsum[r] * alpha[r] + ps;
      pb[(r + 8 * h) * LPS + tl]      = (__bf16)p0;   // C-layout -> LDS
      pb[(r + 8 * h) * LPS + 16 + tl] = (__bf16)p1;
    }
    #pragma unroll
    for (int nt = 0; nt < 16; ++nt) {
      #pragma unroll
      for (int r = 0; r < 8; ++r) acc[nt][r] *= alpha[r];
    }

    // wave-local LDS fence before reading P back in A-layout
    asm volatile("s_wait_dscnt 0" ::: "memory");

    v16bf pf;
    {
      v8bf plo = *(const v8bf*)(pb + tl * LPS + 16 * h);
      v8bf phi = *(const v8bf*)(pb + tl * LPS + 16 * h + 8);
      #pragma unroll
      for (int e = 0; e < 8; ++e) { pf[e] = plo[e]; pf[8 + e] = phi[e]; }
    }

    // O += P @ V   (B operand contiguous from transposed-V LDS tile)
    #pragma unroll
    for (int nt = 0; nt < 16; ++nt) {
      v16bf vb = *(const v16bf*)(vbuf + (size_t)(nt * 16 + tl) * LVS + 16 * h);
      acc[nt] = wmma_bf16(pf, vb, acc[nt]);
    }
  }

  #pragma unroll
  for (int r = 0; r < 8; ++r) rsum[r] = 1.0f / rsum[r];
  for (int nt = 0; nt < 16; ++nt) {
    const int col = nt * 16 + tl;
    #pragma unroll
    for (int r = 0; r < 8; ++r)
      Ob[(size_t)(m0 + r + 8 * h) * DIM + col] = (__bf16)(acc[nt][r] * rsum[r]);
  }
}

// ---------------------------------------------------------------------------
// Kernel 3: out = O @ Wo + bo  (f32 result)
// ---------------------------------------------------------------------------
__global__ __launch_bounds__(256) void out_proj_kernel(
    const __bf16* __restrict__ Ob, const __bf16* __restrict__ WoT,
    const float* __restrict__ bo, float* __restrict__ out)
{
  const int lane = threadIdx.x & 31;
  const int wave = threadIdx.x >> 5;
  const int tl   = lane & 15;
  const int h    = lane >> 4;
  const int m0   = blockIdx.x * 128 + wave * 16;

  __builtin_prefetch(WoT + (size_t)threadIdx.x * 256, 0, 0);

  v16bf a[8];
  const __bf16* orow = Ob + (size_t)(m0 + tl) * DIM;
  #pragma unroll
  for (int dk = 0; dk < 8; ++dk) {
    v8bf lo = *(const v8bf*)(orow + dk * 32 + 16 * h);
    v8bf hi = *(const v8bf*)(orow + dk * 32 + 16 * h + 8);
    #pragma unroll
    for (int e = 0; e < 8; ++e) { a[dk][e] = lo[e]; a[dk][8 + e] = hi[e]; }
  }

  for (int nt = 0; nt < 16; ++nt) {
    v8f c = {};
    const __bf16* w = WoT + (size_t)(nt * 16 + tl) * DIM + 16 * h;
    #pragma unroll
    for (int kt = 0; kt < 8; ++kt)
      c = wmma_bf16(a[kt], *(const v16bf*)(w + kt * 32), c);
    const int col = nt * 16 + tl;
    const float bv = bo[col];
    #pragma unroll
    for (int r = 0; r < 8; ++r)
      out[(size_t)(m0 + r + 8 * h) * DIM + col] = c[r] + bv;
  }
}

// ---------------------------------------------------------------------------
extern "C" void kernel_launch(void* const* d_in, const int* in_sizes, int n_in,
                              void* d_out, int out_size, void* d_ws, size_t ws_size,
                              hipStream_t stream) {
  (void)in_sizes; (void)n_in; (void)out_size; (void)ws_size;
  const float* x  = (const float*)d_in[0];
  const float* Wq = (const float*)d_in[1];
  const float* bq = (const float*)d_in[2];
  const float* Wk = (const float*)d_in[3];
  const float* bk = (const float*)d_in[4];
  const float* Wv = (const float*)d_in[5];
  const float* bv = (const float*)d_in[6];
  const float* Wo = (const float*)d_in[7];
  const float* bo = (const float*)d_in[8];

  char* ws = (char*)d_ws;
  __bf16* WT = (__bf16*)(ws);                                  //  512 KB (4 x 256x256 bf16)
  __bf16* Qb = (__bf16*)(ws + 524288);                         //  8 MB
  __bf16* Kb = (__bf16*)(ws + 524288 + (size_t)8388608);       //  8 MB
  __bf16* Vt = (__bf16*)(ws + 524288 + (size_t)2 * 8388608);   //  8 MB (transposed V)
  __bf16* Ob = (__bf16*)(ws + 524288 + (size_t)3 * 8388608);   //  8 MB

  transpose_weights<<<dim3(256, 4, 1), 256, 0, stream>>>(Wq, Wk, Wv, Wo, WT);
  qkv_proj_kernel<<<MTOT / 128, 256, 0, stream>>>(x, WT, bq, bk, bv, Qb, Kb, Vt);
  flash_attn_kernel<<<MTOT / 128, 256, 0, stream>>>(Qb, Kb, Vt, Ob);
  out_proj_kernel<<<MTOT / 128, 256, 0, stream>>>(Ob, WT + 3 * 65536, bo,
                                                  (float*)d_out);
}
